// Block_21114059227476
// MI455X (gfx1250) — matrix operand order repaired and checked by reading, not compiled, not run on hardware
//
// AFNO block for MI455X (gfx1250), compile-only development.
//
// All contractions -> bf16 WMMA GEMMs (fp32 accum); rfft2/irfft2 realized as
// truncated DFT-matrix GEMMs (91 kept W-modes, full 180 H-modes), norm='ortho'
// folded into coefficient matrices. ~651 GFLOP total, 612 GFLOP in the MLP.
//
// Single-path GEMM: every call is padded at the buffer level (K multiple of
// 32 via zero-padded coefficient columns, A-row over-allocation for M
// remainders, N always multiple of 16), so the wave32 inner loop is fully
// branch-free: 16 coalesced b32 B loads + 16 b128 A loads + 4x
// v_wmma_f32_16x16x32_bf16 per 32-wide K step. Stores keep an m<M guard.

#include <hip/hip_runtime.h>
#include <hip/hip_bf16.h>

typedef __attribute__((ext_vector_type(16))) __bf16 v16bf;
typedef __attribute__((ext_vector_type(8)))  float  v8f;

enum { EP_NONE = 0, EP_GELU = 1, EP_GELU_MOD = 2, EP_RELU_MOD = 3, EP_SHRINK = 4 };

struct GemmP {
  const float* A; const float* B; const float* C; float* D;
  const float* bias; const float* scale; const float* shift;
  const float* res1; const float* res2;
  long sAm, bA;          // A: row stride (K contiguous); rows readable up to 64-aligned M
  long sBk, bB;          // B: K-row stride (N contiguous); rows readable up to K (padded)
  long sDm, bD;          // D: row stride (N contiguous)
  long sRm, sRn, bR;
  int M, N, K;           // K must be a multiple of 32; N a multiple of 16
  float alpha;
  int ep;
};

__device__ __forceinline__ float gelu_f(float x) {
  return 0.5f * x * (1.0f + erff(x * 0.70710678118654752f));
}

__device__ __forceinline__ v16bf pack_a(float4 a0, float4 a1, float4 a2, float4 a3) {
  v16bf f;
  f[0] = (__bf16)a0.x; f[1]  = (__bf16)a0.y; f[2]  = (__bf16)a0.z; f[3]  = (__bf16)a0.w;
  f[4] = (__bf16)a1.x; f[5]  = (__bf16)a1.y; f[6]  = (__bf16)a1.z; f[7]  = (__bf16)a1.w;
  f[8] = (__bf16)a2.x; f[9]  = (__bf16)a2.y; f[10] = (__bf16)a2.z; f[11] = (__bf16)a2.w;
  f[12] = (__bf16)a3.x; f[13] = (__bf16)a3.y; f[14] = (__bf16)a3.z; f[15] = (__bf16)a3.w;
  return f;
}

// A fragment (16x32 CDNA5 layout): lanes 0-15 hold row M=l16, K {0..7,16..23};
// lanes 16-31 K {8..15,24..31}. B fragment (32x16): lane = column n=l16,
// same K order as A (half 0 -> K {0..7,16..23}, half 1 -> K {8..15,24..31}). C/D: lane n, VGPR v -> row half*8+v.
__global__ void gemm_wmma(GemmP p) {
  const int wave = (int)(threadIdx.x >> 5);
  const int lane = (int)(threadIdx.x & 31);
  const int half = lane >> 4;
  const int l16  = lane & 15;
  const long ntn  = (long)(p.N >> 4);
  const long ntmw = (long)((p.M + 63) >> 6);
  const long tid  = (long)blockIdx.x * 8 + wave;
  if (tid >= ntmw * ntn) return;                 // wave-uniform exit
  const int  tn = (int)(tid % ntn);
  const int  m0 = (int)(tid / ntn) * 64;
  const long batch = (long)blockIdx.y;
  const float* Ab = p.A + p.bA * batch;
  const float* Bb = p.B + p.bB * batch;
  const int M = p.M, K = p.K;
  const long sAm = p.sAm, sBk = p.sBk;
  const int nB = tn * 16 + l16;

  v8f acc[4];
#pragma unroll
  for (int t = 0; t < 4; ++t) acc[t] = (v8f){0.f,0.f,0.f,0.f,0.f,0.f,0.f,0.f};

  for (int k0 = 0; k0 < K; k0 += 32) {
    const float* Bp = Bb + (long)(k0 + half * 8) * sBk + nB;
    float bv[16];
#pragma unroll
    for (int i = 0; i < 16; ++i) bv[i] = Bp[(long)(i < 8 ? i : i + 8) * sBk];
    v16bf bf;
#pragma unroll
    for (int i = 0; i < 16; ++i) bf[i] = (__bf16)bv[i];
    __builtin_prefetch(Bp + 32L * sBk, 0, 1);

#pragma unroll
    for (int t = 0; t < 4; ++t) {
      const float* Ap = Ab + (long)(m0 + t * 16 + l16) * sAm + k0 + half * 8;
      float4 a0 = *(const float4*)(Ap);
      float4 a1 = *(const float4*)(Ap + 4);
      float4 a2 = *(const float4*)(Ap + 16);
      float4 a3 = *(const float4*)(Ap + 20);
      __builtin_prefetch(Ap + 32, 0, 1);
      v16bf af = pack_a(a0, a1, a2, a3);
      acc[t] = __builtin_amdgcn_wmma_f32_16x16x32_bf16(false, af, false, bf,
                                                       (short)0, acc[t], false, false);
    }
  }

#pragma unroll
  for (int t = 0; t < 4; ++t) {
#pragma unroll
    for (int r = 0; r < 8; ++r) {
      int m = m0 + t * 16 + half * 8 + r;
      if (m >= M) continue;
      long di = (long)m * p.sDm + nB + p.bD * batch;
      float x = p.alpha * acc[t][r];
      if (p.C)    x += p.C[di];
      if (p.bias) x += p.bias[nB];
      if      (p.ep == EP_GELU)     x = gelu_f(x);
      else if (p.ep == EP_GELU_MOD) x = gelu_f(x * (1.0f + p.scale[nB]) + p.shift[nB]);
      else if (p.ep == EP_RELU_MOD) { x = x * (1.0f + p.scale[nB]) + p.shift[nB]; x = fmaxf(x, 0.0f); }
      else if (p.ep == EP_SHRINK)   { x = (x > 0.01f) ? (x - 0.01f) : ((x < -0.01f) ? (x + 0.01f) : 0.0f); }
      if (p.res1 || p.res2) {
        long ri = (long)m * p.sRm + (long)nB * p.sRn + p.bR * batch;
        if (p.res1) x += p.res1[ri];
        if (p.res2) x += p.res2[ri];
      }
      p.D[di] = x;
    }
  }
}

// Row LayerNorm over C=768, one block (256 threads) per row.
__global__ void ln_kernel(const float* __restrict__ x, const float* __restrict__ w,
                          const float* __restrict__ bvec, float* __restrict__ out, int C) {
  __shared__ float red[256];
  const long row = (long)blockIdx.x;
  const float* xr = x + row * C;
  float s = 0.f;
  for (int c = threadIdx.x; c < C; c += 256) s += xr[c];
  red[threadIdx.x] = s; __syncthreads();
  for (int off = 128; off > 0; off >>= 1) {
    if ((int)threadIdx.x < off) red[threadIdx.x] += red[threadIdx.x + off];
    __syncthreads();
  }
  const float mean = red[0] / (float)C;
  __syncthreads();
  float v = 0.f;
  for (int c = threadIdx.x; c < C; c += 256) { float d = xr[c] - mean; v += d * d; }
  red[threadIdx.x] = v; __syncthreads();
  for (int off = 128; off > 0; off >>= 1) {
    if ((int)threadIdx.x < off) red[threadIdx.x] += red[threadIdx.x + off];
    __syncthreads();
  }
  const float rstd = rsqrtf(red[0] / (float)C + 1e-5f);
  __syncthreads();
  for (int c = threadIdx.x; c < 768; c += 256)
    out[row * C + c] = (xr[c] - mean) * rstd * w[c] + bvec[c];
}

// DFT coefficient matrices, ortho norm (1/sqrt(64800)) folded into forward-W
// and inverse-W matrices. Angles reduced exactly via integer modulo.
// All matrices padded: CW/SW 128x384 (valid 91x360), CH/SH 192x192 (valid
// 180x180), EC/ES 384x96 (valid 360x91), e_pad 64x64 (row 0 = mod_embed).
__global__ void init_mats(float* CW, float* SW, float* CH, float* SH,
                          float* EC, float* ES, float* e_pad, const float* e) {
  const int i = blockIdx.x * 256 + threadIdx.x;
  const float norm = rsqrtf(64800.0f);
  const float TWO_PI = 6.28318530717958647692f;
  if (i < 128 * 384) {              // forward W-DFT (rows w', cols w)
    int wp = i / 384, w = i % 384;
    float c = 0.f, s = 0.f;
    if (wp < 91 && w < 360) {
      int r = (wp * w) % 360;
      float th = TWO_PI * (float)r / 360.0f;
      c = cosf(th) * norm; s = -sinf(th) * norm;
    }
    CW[i] = c; SW[i] = s;
  }
  if (i < 192 * 192) {              // H-DFT (symmetric, shared fwd/inv)
    int a = i / 192, b = i % 192;
    float c = 0.f, s = 0.f;
    if (a < 180 && b < 180) {
      int r = (a * b) % 180;
      float th = TWO_PI * (float)r / 180.0f;
      c = cosf(th); s = sinf(th);
    }
    CH[i] = c; SH[i] = s;
  }
  if (i < 384 * 96) {               // inverse W-DFT (real), hermitian weights
    int w = i / 96, wp = i % 96;
    float c = 0.f, s = 0.f;
    if (wp < 91 && w < 360) {
      float g = (wp == 0) ? 1.0f : 2.0f;
      int r = (w * wp) % 360;
      float th = TWO_PI * (float)r / 360.0f;
      c = g * cosf(th) * norm; s = -g * sinf(th) * norm;
    }
    EC[i] = c; ES[i] = s;
  }
  if (i < 64 * 64) e_pad[i] = (i < 64) ? e[i] : 0.0f;
}

static GemmP mk(const float* A, long sAm, long bA,
                const float* B, long sBk, long bB,
                float* D, long sDm, long bD,
                int M, int N, int K, float alpha, int ep,
                const float* Cc = nullptr, const float* bias = nullptr,
                const float* scale = nullptr, const float* shift = nullptr,
                const float* r1 = nullptr, const float* r2 = nullptr,
                long sRm = 0, long sRn = 0, long bR = 0) {
  GemmP p;
  p.A = A; p.B = B; p.C = Cc; p.D = D;
  p.bias = bias; p.scale = scale; p.shift = shift; p.res1 = r1; p.res2 = r2;
  p.sAm = sAm; p.bA = bA;
  p.sBk = sBk; p.bB = bB;
  p.sDm = sDm; p.bD = bD;
  p.sRm = sRm; p.sRn = sRn; p.bR = bR;
  p.M = M; p.N = N; p.K = K; p.alpha = alpha; p.ep = ep;
  return p;
}

static void run(hipStream_t st, const GemmP& p, int batch) {
  long tiles = (long)((p.M + 63) / 64) * (long)(p.N / 16);
  dim3 grid((unsigned)((tiles + 7) / 8), (unsigned)batch, 1);
  gemm_wmma<<<grid, 256, 0, st>>>(p);
}

extern "C" void kernel_launch(void* const* d_in, const int* in_sizes, int n_in,
                              void* d_out, int out_size, void* d_ws, size_t ws_size,
                              hipStream_t stream) {
  (void)in_sizes; (void)n_in; (void)out_size; (void)ws_size;

  const float* x     = (const float*)d_in[0];
  const float* e     = (const float*)d_in[1];
  const float* n1w   = (const float*)d_in[2];
  const float* n1b   = (const float*)d_in[3];
  const float* n2w   = (const float*)d_in[4];
  const float* n2b   = (const float*)d_in[5];
  const float* w1    = (const float*)d_in[6];   // (2,8,96,96)
  const float* b1    = (const float*)d_in[7];   // (2,8,96)
  const float* w2    = (const float*)d_in[8];
  const float* b2    = (const float*)d_in[9];
  const float* fs_w0 = (const float*)d_in[10];  // (64,1536)
  const float* fs_b0 = (const float*)d_in[11];
  const float* fs_w1 = (const float*)d_in[12];  // (1536,1536)
  const float* fs_b1 = (const float*)d_in[13];
  const float* fc1w  = (const float*)d_in[14];  // (768,3072)
  const float* fc1b  = (const float*)d_in[15];
  const float* fc2w  = (const float*)d_in[16];  // (3072,768)
  const float* fc2b  = (const float*)d_in[17];
  const float* ms_w0 = (const float*)d_in[18];  // (64,6144)
  const float* ms_b0 = (const float*)d_in[19];
  const float* ms_w1 = (const float*)d_in[20];  // (6144,6144)
  const float* ms_b1 = (const float*)d_in[21];
  float* out = (float*)d_out;

  const long HWC  = 180L * 360 * 768;   // 49,766,400
  const long SPEC = 180L * 96 * 768;    // 13,271,040
  const long ROWW = 360L * 768;         // 276,480 (per-h stride, spatial)
  const long ROWF = 96L * 768;          // 73,728  (per-h stride, spectral)

  float* w = (float*)d_ws;
  long o = 0;
  auto alloc = [&](long n) { float* p = w + o; o += n; return p; };
  float* xln  = alloc(HWC);
  float* P1   = alloc(SPEC);            // Frw -> o1r -> Hr
  float* P2   = alloc(SPEC);            // Fiw -> o1i -> Hi
  float* P3   = alloc(SPEC);            // Gr  -> o2r
  float* P4   = alloc(SPEC);            // Gi  -> o2i
  float* h1   = alloc(HWC);
  float* xln2 = alloc(HWC);
  float* hmid = alloc(64800L * 3072);
  float* CW = alloc(128L * 384);        // padded 91x360 forward-W cos
  float* SW = alloc(128L * 384);        // padded forward-W sin
  float* CH = alloc(192L * 192);        // padded 180x180 H cos
  float* SH = alloc(192L * 192);        // padded H sin
  float* EC = alloc(384L * 96);         // padded 360x91 inverse-W cos
  float* ES = alloc(384L * 96);         // padded inverse-W sin
  float* e_pad = alloc(64L * 64);       // row 0 = mod_embed, rest 0
  float* g1  = alloc(64L * 1536);       // row 0 live, rest over-read slack
  float* sb1 = alloc(1536);
  float* g2  = alloc(64L * 6144);       // row 0 live, rest over-read slack
  float* sb2 = alloc(6144);
  (void)alloc(64L * 6144);              // tail slack for A-row over-reads

  init_mats<<<192, 256, 0, stream>>>(CW, SW, CH, SH, EC, ES, e_pad, e);

  // LN1
  ln_kernel<<<64800, 256, 0, stream>>>(x, n1w, n1b, xln, 768);

  // Forward DFT along W (batched over h): F[h,w',c].  K padded 360->384.
  run(stream, mk(CW, 384, 0, xln, 768, ROWW, P1, 768, ROWF, 96, 768, 384, 1.f, EP_NONE), 180);
  run(stream, mk(SW, 384, 0, xln, 768, ROWW, P2, 768, ROWF, 96, 768, 384, 1.f, EP_NONE), 180);

  // Forward DFT along H (complex): Gr = CH@Fr + SH@Fi ; Gi = CH@Fi - SH@Fr.
  // K padded 180->192 (zero coefficient columns; B over-reads stay in ws).
  run(stream, mk(CH, 192, 0, P1, ROWF, 0, P3, ROWF, 0, 180, (int)ROWF, 192,  1.f, EP_NONE), 1);
  run(stream, mk(SH, 192, 0, P2, ROWF, 0, P3, ROWF, 0, 180, (int)ROWF, 192,  1.f, EP_NONE, P3), 1);
  run(stream, mk(CH, 192, 0, P2, ROWF, 0, P4, ROWF, 0, 180, (int)ROWF, 192,  1.f, EP_NONE), 1);
  run(stream, mk(SH, 192, 0, P1, ROWF, 0, P4, ROWF, 0, 180, (int)ROWF, 192, -1.f, EP_NONE, P4), 1);

  // Filter scale/shift: sb1 = gelu(e@fs_w0+fs_b0)@fs_w1 + fs_b1
  run(stream, mk(e_pad, 64, 0, fs_w0, 1536, 0, g1, 1536, 0, 1, 1536, 64,   1.f, EP_GELU, nullptr, fs_b0), 1);
  run(stream, mk(g1, 1536, 0, fs_w1, 1536, 0, sb1, 1536, 0, 1, 1536, 1536, 1.f, EP_NONE, nullptr, fs_b1), 1);

  // Block-diagonal complex 2-layer spectral MLP (positions M = 180*96 = 17280)
  for (int b = 0; b < 8; ++b) {
    const float* W1r = w1 + (long)b * 9216;
    const float* W1i = w1 + (long)(8 + b) * 9216;
    const float* W2r = w2 + (long)b * 9216;
    const float* W2i = w2 + (long)(8 + b) * 9216;
    const float* b1r = b1 + (long)b * 96;
    const float* b1i = b1 + (long)(8 + b) * 96;
    const float* b2r = b2 + (long)b * 96;
    const float* b2i = b2 + (long)(8 + b) * 96;
    float* Grb = P3 + 96L * b;  float* Gib = P4 + 96L * b;
    float* o1r = P1 + 96L * b;  float* o1i = P2 + 96L * b;

    // o1r = relu((Gr@W1r - Gi@W1i + b1r)*(1+s) + t)
    run(stream, mk(Grb, 768, 0, W1r, 96, 0, o1r, 768, 0, 17280, 96, 96,  1.f, EP_NONE), 1);
    run(stream, mk(Gib, 768, 0, W1i, 96, 0, o1r, 768, 0, 17280, 96, 96, -1.f, EP_RELU_MOD,
                   o1r, b1r, sb1 + 96 * b, sb1 + 768 + 96 * b), 1);
    // o1i = relu((Gi@W1r + Gr@W1i + b1i)*(1+s) + t)
    run(stream, mk(Gib, 768, 0, W1r, 96, 0, o1i, 768, 0, 17280, 96, 96,  1.f, EP_NONE), 1);
    run(stream, mk(Grb, 768, 0, W1i, 96, 0, o1i, 768, 0, 17280, 96, 96,  1.f, EP_RELU_MOD,
                   o1i, b1i, sb1 + 96 * b, sb1 + 768 + 96 * b), 1);
    // o2r = shrink(o1r@W2r - o1i@W2i + b2r)    (written back over Gr)
    run(stream, mk(o1r, 768, 0, W2r, 96, 0, Grb, 768, 0, 17280, 96, 96,  1.f, EP_NONE), 1);
    run(stream, mk(o1i, 768, 0, W2i, 96, 0, Grb, 768, 0, 17280, 96, 96, -1.f, EP_SHRINK, Grb, b2r), 1);
    // o2i = shrink(o1i@W2r + o1r@W2i + b2i)    (written back over Gi)
    run(stream, mk(o1i, 768, 0, W2r, 96, 0, Gib, 768, 0, 17280, 96, 96,  1.f, EP_NONE), 1);
    run(stream, mk(o1r, 768, 0, W2i, 96, 0, Gib, 768, 0, 17280, 96, 96,  1.f, EP_SHRINK, Gib, b2i), 1);
  }

  // Inverse DFT along H: Hr = CH@o2r - SH@o2i ; Hi = CH@o2i + SH@o2r
  run(stream, mk(CH, 192, 0, P3, ROWF, 0, P1, ROWF, 0, 180, (int)ROWF, 192,  1.f, EP_NONE), 1);
  run(stream, mk(SH, 192, 0, P4, ROWF, 0, P1, ROWF, 0, 180, (int)ROWF, 192, -1.f, EP_NONE, P1), 1);
  run(stream, mk(CH, 192, 0, P4, ROWF, 0, P2, ROWF, 0, 180, (int)ROWF, 192,  1.f, EP_NONE), 1);
  run(stream, mk(SH, 192, 0, P3, ROWF, 0, P2, ROWF, 0, 180, (int)ROWF, 192,  1.f, EP_NONE, P2), 1);

  // Inverse real DFT along W (batched over h) + filter bias (xln) + residual (x)
  run(stream, mk(EC, 96, 0, P1, 768, ROWF, h1, 768, ROWW, 360, 768, 96, 1.f, EP_NONE), 180);
  run(stream, mk(ES, 96, 0, P2, 768, ROWF, h1, 768, ROWW, 360, 768, 96, 1.f, EP_NONE,
                 h1, nullptr, nullptr, nullptr, xln, x, 768, 1, ROWW), 180);

  // LN2
  ln_kernel<<<64800, 256, 0, stream>>>(h1, n2w, n2b, xln2, 768);

  // MLP scale/shift: sb2 = gelu(e@ms_w0+ms_b0)@ms_w1 + ms_b1
  run(stream, mk(e_pad, 64, 0, ms_w0, 6144, 0, g2, 6144, 0, 1, 6144, 64,   1.f, EP_GELU, nullptr, ms_b0), 1);
  run(stream, mk(g2, 6144, 0, ms_w1, 6144, 0, sb2, 6144, 0, 1, 6144, 6144, 1.f, EP_NONE, nullptr, ms_b1), 1);

  // MLP: hmid = gelu((xln2@fc1w + fc1b)*(1+s2) + t2)
  run(stream, mk(xln2, 768, 0, fc1w, 3072, 0, hmid, 3072, 0, 64800, 3072, 768,
                 1.f, EP_GELU_MOD, nullptr, fc1b, sb2, sb2 + 3072), 1);
  // out = hmid@fc2w + fc2b + h1
  run(stream, mk(hmid, 3072, 0, fc2w, 768, 0, out, 768, 0, 64800, 768, 3072,
                 1.f, EP_NONE, nullptr, fc2b, nullptr, nullptr, h1, nullptr, 768, 1, 0), 1);
}
